// MaskedAttention3D_824633721340
// MI455X (gfx1250) — matrix operand order repaired and checked by reading, not compiled, not run on hardware
//
#include <hip/hip_runtime.h>
#include <hip/hip_bf16.h>
#include <stdint.h>

typedef __bf16 bf16_t;
typedef __attribute__((ext_vector_type(16))) __bf16 v16bf;
typedef __attribute__((ext_vector_type(8)))  __bf16 v8bf;
typedef __attribute__((ext_vector_type(8)))  float  v8f;

#define B_      2
#define NTOK    32768          // 32*32*32
#define CDIM    528
#define KPAD    544            // 17 * 32
#define KSTEPS  17
#define NHEAD   8
#define HD      66
#define HDP     96             // head_dim padded to 3*32
#define NQ      100
#define NQP     112            // 7 * 16
#define NKV     1056           // 2*CDIM
#define NKVP    1088           // 17 * 64

// ---------- helpers ----------

static __device__ __forceinline__ bf16_t f2bf(float f) {
    union { float f; uint32_t u; } x; x.f = f;
    uint32_t r = x.u + 0x7FFFu + ((x.u >> 16) & 1u);   // round to nearest even
    union { uint16_t u; bf16_t b; } y; y.u = (uint16_t)(r >> 16);
    return y.b;
}

static __device__ __forceinline__ v8f wmma_bf16(v16bf a, v16bf b, v8f c) {
    return __builtin_amdgcn_wmma_f32_16x16x32_bf16(
        false, a, false, b, (short)0, c, false, false);
}

// A-fragment (16x32 bf16, M x K): lane row = m, chunks K=[8*hf,+8) and K=[16+8*hf,+8)
static __device__ __forceinline__ v16bf load_a_frag(const bf16_t* rowk, int hf) {
    union { v16bf v; v8bf h[2]; } u;
    u.h[0] = *(const v8bf*)(rowk + 8 * hf);
    u.h[1] = *(const v8bf*)(rowk + 16 + 8 * hf);
    return u.v;
}

// B-fragment (32x16 bf16, K x N): lane col = n, chunks K=[8*hf,+8) and K=[16+8*hf,+8)
static __device__ __forceinline__ v16bf load_b_frag(const bf16_t* rowk, int hf) {
    union { v16bf v; v8bf h[2]; } u;
    u.h[0] = *(const v8bf*)(rowk + 8 * hf);
    u.h[1] = *(const v8bf*)(rowk + 16 + 8 * hf);
    return u.v;
}

// Async global->LDS 16-byte copy (CDNA5 GLOBAL_LOAD_ASYNC_TO_LDS_B128, ASYNCcnt)
static __device__ __forceinline__ void async_b128(uint32_t lds_addr, const void* gptr) {
    asm volatile("global_load_async_to_lds_b128 %0, %1, off"
                 :: "v"(lds_addr), "v"(gptr) : "memory");
}
static __device__ __forceinline__ uint32_t lds_lo32(const void* p) {
    return (uint32_t)(uintptr_t)p;      // generic addr [31:0] == LDS offset
}
static __device__ __forceinline__ void wait_async(int n) {
    if (n == 0)      asm volatile("s_wait_asynccnt 0x0" ::: "memory");
    else if (n == 2) asm volatile("s_wait_asynccnt 0x2" ::: "memory");
    else             asm volatile("s_wait_asynccnt 0x3" ::: "memory");
}

// ---------- prep kernels ----------

__global__ void zero_kernel(uint32_t* p, size_t ndw) {
    size_t i = (size_t)blockIdx.x * blockDim.x + threadIdx.x;
    size_t step = (size_t)gridDim.x * blockDim.x;
    for (; i < ndw; i += step) p[i] = 0u;
}

__global__ void cvt_pad_kernel(const float* __restrict__ src, bf16_t* __restrict__ dst,
                               int rows, int scols, int dcols) {
    size_t idx  = (size_t)blockIdx.x * blockDim.x + threadIdx.x;
    size_t step = (size_t)gridDim.x * blockDim.x;
    size_t total = (size_t)rows * dcols;
    for (; idx < total; idx += step) {
        int c = (int)(idx % dcols);
        size_t r = idx / dcols;
        dst[idx] = (c < scols) ? f2bf(src[r * scols + c]) : f2bf(0.0f);
    }
}

// qb[h][q][d] = query_embed[q][h*66+d], padded q->112, d->96
__global__ void qb_prep_kernel(const float* __restrict__ qe, bf16_t* __restrict__ qb) {
    int idx  = blockIdx.x * blockDim.x + threadIdx.x;
    int step = gridDim.x * blockDim.x;
    int total = NHEAD * NQP * HDP;
    for (; idx < total; idx += step) {
        int d = idx % HDP;
        int q = (idx / HDP) % NQP;
        int h = idx / (HDP * NQP);
        float v = (q < NQ && d < HD) ? qe[q * CDIM + h * HD + d] : 0.0f;
        qb[idx] = f2bf(v);
    }
}

__global__ void attnb_prep_kernel(const float* __restrict__ attn, bf16_t* __restrict__ attnb) {
    int idx  = blockIdx.x * blockDim.x + threadIdx.x;
    int step = gridDim.x * blockDim.x;
    int total = B_ * NQP * KPAD;
    for (; idx < total; idx += step) {
        int c = idx % KPAD;
        int m = idx / KPAD;
        int b = m / NQP, q = m % NQP;
        float v = (q < NQ && c < CDIM) ? attn[((size_t)b * NQ + q) * CDIM + c] : 0.0f;
        attnb[idx] = f2bf(v);
    }
}

// ---------- kernel 1: kv projection GEMM (65536 x 1056, K=544) ----------
// 256 thr (8 waves); block tile 64(M) x 64(N); wave tile 32(M) x 16(N).
// A and B k-slices staged in LDS via double-buffered async copies.
__global__ void kv_gemm_kernel(const bf16_t* __restrict__ xb, const bf16_t* __restrict__ wkv,
                               const float* __restrict__ kv_b,
                               bf16_t* __restrict__ kb, bf16_t* __restrict__ vbt) {
    __shared__ bf16_t shA[2][64 * 32];   // [buf][local_m][k32]
    __shared__ bf16_t shB[2][64 * 32];   // [buf][local_n][k32]

    int tn = blockIdx.x;                 // 0..16
    int tm = blockIdx.y;                 // 0..1023
    int t = threadIdx.x;
    int wave = t >> 5, lane = t & 31;
    int wm = wave >> 2, wn = wave & 3;
    int hf = lane >> 4, ln = lane & 15;

    int m_blk = tm * 64, n_blk = tn * 64;

    // async copier mapping: 256 threads x 16B cover one 64x32 bf16 tile
    int crow = t >> 2, cch = (t & 3) * 8;          // row 0..63, k-offset 0/8/16/24
    const bf16_t* gA = xb  + (size_t)(m_blk + crow) * KPAD + cch;
    const bf16_t* gB = wkv + (size_t)(n_blk + crow) * KPAD + cch;
    uint32_t ldsA[2], ldsB[2];
    for (int bfi = 0; bfi < 2; ++bfi) {
        ldsA[bfi] = lds_lo32(&shA[bfi][crow * 32 + cch]);
        ldsB[bfi] = lds_lo32(&shB[bfi][crow * 32 + cch]);
    }

    v8f acc0, acc1;
    for (int i = 0; i < 8; ++i) { acc0[i] = 0.f; acc1[i] = 0.f; }

    // prologue: stage k-slice 0 into buffer 0
    async_b128(ldsA[0], gA);
    async_b128(ldsB[0], gB);

    for (int ks = 0; ks < KSTEPS; ++ks) {
        int cur = ks & 1;
        if (ks + 1 < KSTEPS) {           // prefetch next k-slice into other buffer
            async_b128(ldsA[1 - cur], gA + (ks + 1) * 32);
            async_b128(ldsB[1 - cur], gB + (ks + 1) * 32);
            wait_async(2);               // older (current-buffer) copies complete, in order
        } else {
            wait_async(0);
        }
        __syncthreads();                 // all waves' slices visible in LDS

        const bf16_t* Ab = &shA[cur][0];
        const bf16_t* Bb = &shB[cur][0];
        v16bf a0 = load_a_frag(Ab + (wm * 32 + ln) * 32, hf);
        v16bf a1 = load_a_frag(Ab + (wm * 32 + 16 + ln) * 32, hf);
        v16bf b  = load_b_frag(Bb + (wn * 16 + ln) * 32, hf);
        acc0 = wmma_bf16(a0, b, acc0);
        acc1 = wmma_bf16(a1, b, acc1);

        __syncthreads();                 // reads done before buffer is overwritten
    }

    int m0 = m_blk + wm * 32;
    int nc = n_blk + wn * 16 + ln;       // output column (lane-invariant over i)
    if (nc >= NKV) return;
    float bias = kv_b[nc];
    int is_v = nc >= CDIM;
    int c2 = is_v ? nc - CDIM : nc;
    int h = c2 / HD, d = c2 % HD;

    for (int half = 0; half < 2; ++half) {
        v8f acc = half ? acc1 : acc0;
        int mb = m0 + half * 16 + 8 * hf;
        for (int i = 0; i < 8; ++i) {
            int m = mb + i;
            int b = m >> 15, nt = m & (NTOK - 1);
            bf16_t val = f2bf(acc[i] + bias);
            size_t bh = (size_t)b * NHEAD + h;
            if (!is_v) kb [(bh * NTOK + nt) * HDP + d] = val;    // K: [bh][n][96]
            else       vbt[(bh * HDP + d) * NTOK + nt] = val;    // V^T: [bh][96][n]
        }
    }
}

// ---------- kernel 2: scores S[bh][q][n] = scale * q . k ----------
__global__ void scores_kernel(const bf16_t* __restrict__ qb, const bf16_t* __restrict__ kb,
                              float* __restrict__ sc) {
    int chunk = blockIdx.x;              // 1024 keys
    int bh = blockIdx.y;
    int wave = threadIdx.x >> 5, lane = threadIdx.x & 31;
    if (wave >= 7) return;               // no barriers in this kernel
    int hf = lane >> 4, ln = lane & 15;
    int h = bh & 7;

    const bf16_t* qrow = qb + ((size_t)h * NQP + wave * 16 + ln) * HDP;
    v16bf qf[3];
    for (int kk = 0; kk < 3; ++kk) qf[kk] = load_a_frag(qrow + kk * 32, hf);

    const bf16_t* kbase = kb + (size_t)bh * NTOK * HDP;
    float* srow = sc + ((size_t)bh * NQP + wave * 16) * NTOK;
    const float scale = 0.12309149097933274f;   // 66^-0.5
    int key0 = chunk * 1024;

    for (int s = 0; s < 64; ++s) {
        int n0 = key0 + s * 16;
        const bf16_t* krow = kbase + (size_t)(n0 + ln) * HDP;
        v8f acc; for (int i = 0; i < 8; ++i) acc[i] = 0.f;
        for (int kk = 0; kk < 3; ++kk)
            acc = wmma_bf16(qf[kk], load_b_frag(krow + kk * 32, hf), acc);
        int col = n0 + ln;
        for (int i = 0; i < 8; ++i)
            srow[(size_t)(8 * hf + i) * NTOK + col] = acc[i] * scale;
    }
}

// ---------- kernel 3: masked softmax over keys, write P in bf16 ----------
__global__ void softmax_kernel(const float* __restrict__ sc, const int* __restrict__ mask,
                               bf16_t* __restrict__ pb) {
    int r = blockIdx.x;                  // bh*112 + q
    int bh = r / NQP, q = r % NQP;
    int b = bh >> 3;
    bf16_t* prow = pb + (size_t)r * NTOK;
    int t = threadIdx.x;

    if (q >= NQ) {                       // pad rows of P must be zero for AV GEMM
        for (int i = t; i < NTOK; i += 256) prow[i] = f2bf(0.0f);
        return;
    }
    const float* row = sc + (size_t)r * NTOK;
    const int* mrow = mask + (size_t)b * NTOK;
    __shared__ float red[256];

    float mx = -1e30f;
    for (int i = t; i < NTOK; i += 256)
        if (mrow[i] != 0) mx = fmaxf(mx, row[i]);
    red[t] = mx; __syncthreads();
    for (int s = 128; s > 0; s >>= 1) { if (t < s) red[t] = fmaxf(red[t], red[t + s]); __syncthreads(); }
    mx = red[0]; __syncthreads();

    float sum = 0.f;
    for (int i = t; i < NTOK; i += 256)
        if (mrow[i] != 0) sum += __expf(row[i] - mx);
    red[t] = sum; __syncthreads();
    for (int s = 128; s > 0; s >>= 1) { if (t < s) red[t] += red[t + s]; __syncthreads(); }
    float inv = 1.0f / red[0];

    for (int i = t; i < NTOK; i += 256)
        prow[i] = f2bf((mrow[i] != 0) ? __expf(row[i] - mx) * inv : 0.0f);
}

// ---------- kernel 4: AV = P x V^T, split-K over 32 key chunks, atomic combine ----------
// V super-tile (96 x 64 keys, 12KB) staged in LDS via double-buffered async copies
// and shared by all 7 compute waves (was 7x redundant VMEM traffic).
__global__ void av_kernel(const bf16_t* __restrict__ pb, const bf16_t* __restrict__ vbt,
                          float* __restrict__ attn) {
    __shared__ bf16_t shV[2][HDP * 64];  // [buf][d][key64]

    int chunk = blockIdx.x;              // 1024 keys
    int bh = blockIdx.y;
    int t = threadIdx.x;
    int wave = t >> 5, lane = t & 31;
    int hf = lane >> 4, ln = lane & 15;
    int b = bh >> 3, h = bh & 7;
    bool active = wave < 7;              // wave 7 only participates in copies/barriers

    const bf16_t* prow  = pb + ((size_t)bh * NQP + (active ? wave * 16 + ln : 0)) * NTOK;
    const bf16_t* vbase = vbt + (size_t)bh * HDP * NTOK;
    int key0 = chunk * 1024;

    v8f acc[6];
    for (int j = 0; j < 6; ++j) for (int i = 0; i < 8; ++i) acc[j][i] = 0.f;

    // copier: 96 rows x 128B = 768 x 16B chunks; each thread issues 3 async b128s
    int c0 = t;
    // prologue: super-step 0 into buffer 0
    for (int r = 0; r < 3; ++r) {
        int c = c0 + 256 * r, row = c >> 3, part = (c & 7) * 8;
        async_b128(lds_lo32(&shV[0][row * 64 + part]),
                   vbase + (size_t)row * NTOK + key0 + part);
    }

    for (int ss = 0; ss < 16; ++ss) {    // 16 super-steps of 64 keys
        int cur = ss & 1;
        if (ss + 1 < 16) {
            int kbase = key0 + (ss + 1) * 64;
            for (int r = 0; r < 3; ++r) {
                int c = c0 + 256 * r, row = c >> 3, part = (c & 7) * 8;
                async_b128(lds_lo32(&shV[1 - cur][row * 64 + part]),
                           vbase + (size_t)row * NTOK + kbase + part);
            }
            wait_async(3);
        } else {
            wait_async(0);
        }
        __syncthreads();

        if (active) {
            for (int kk = 0; kk < 2; ++kk) {
                int k0 = key0 + ss * 64 + kk * 32;
                v16bf a = load_a_frag(prow + k0, hf);
                for (int j = 0; j < 6; ++j) {
                    const bf16_t* vr = &shV[cur][(j * 16 + ln) * 64 + kk * 32];
                    acc[j] = wmma_bf16(a, load_b_frag(vr, hf), acc[j]);
                }
            }
        }
        __syncthreads();
    }

    if (!active) return;
    for (int j = 0; j < 6; ++j) {
        int d = j * 16 + ln;
        if (d >= HD) continue;
        for (int i = 0; i < 8; ++i) {
            int q = wave * 16 + 8 * hf + i;
            if (q >= NQ) continue;
            atomicAdd(&attn[((size_t)b * NQ + q) * CDIM + h * HD + d], acc[j][i]);
        }
    }
}

// ---------- kernel 5: output projection (224 x 528, K=544) ----------
__global__ void proj_gemm_kernel(const bf16_t* __restrict__ attnb, const bf16_t* __restrict__ wpj,
                                 const float* __restrict__ proj_b, float* __restrict__ out) {
    int wg = blockIdx.x * 8 + (threadIdx.x >> 5);   // one 16x16 tile per wave
    const int MT = (B_ * NQP) / 16;                 // 14
    const int NT = CDIM / 16;                       // 33
    if (wg >= MT * NT) return;
    int tm = wg / NT, tn = wg % NT;
    int lane = threadIdx.x & 31;
    int hf = lane >> 4, ln = lane & 15;

    const bf16_t* arow = attnb + (size_t)(tm * 16 + ln) * KPAD;
    const bf16_t* brow = wpj   + (size_t)(tn * 16 + ln) * KPAD;

    v8f acc; for (int i = 0; i < 8; ++i) acc[i] = 0.f;
    for (int k0 = 0; k0 < KPAD; k0 += 32)
        acc = wmma_bf16(load_a_frag(arow + k0, hf), load_b_frag(brow + k0, hf), acc);

    int c = tn * 16 + ln;
    float bias = proj_b[c];
    for (int i = 0; i < 8; ++i) {
        int m = tm * 16 + 8 * hf + i;
        int b = m / NQP, q = m % NQP;
        if (q < NQ)
            out[((size_t)b * NQ + q) * CDIM + c] = acc[i] + bias;
    }
}

// ---------- host launcher ----------

extern "C" void kernel_launch(void* const* d_in, const int* in_sizes, int n_in,
                              void* d_out, int out_size, void* d_ws, size_t ws_size,
                              hipStream_t stream) {
    const float* x     = (const float*)d_in[0];
    const int*   mask  = (const int*)d_in[1];
    const float* qe    = (const float*)d_in[2];
    const float* kvw   = (const float*)d_in[3];
    const float* kvb   = (const float*)d_in[4];
    const float* pw    = (const float*)d_in[5];
    const float* pbias = (const float*)d_in[6];
    float* out = (float*)d_out;
    (void)in_sizes; (void)n_in; (void)out_size; (void)ws_size;

    char* ws = (char*)d_ws;
    size_t off = 0;
    auto alloc = [&](size_t bytes) -> char* {
        char* p = ws + off;
        off += (bytes + 255) & ~(size_t)255;
        return p;
    };
    bf16_t* xb    = (bf16_t*)alloc((size_t)B_ * NTOK * KPAD * 2);
    bf16_t* wkv   = (bf16_t*)alloc((size_t)NKVP * KPAD * 2);
    bf16_t* qb    = (bf16_t*)alloc((size_t)NHEAD * NQP * HDP * 2);
    bf16_t* wpj   = (bf16_t*)alloc((size_t)CDIM * KPAD * 2);
    bf16_t* kb    = (bf16_t*)alloc((size_t)B_ * NHEAD * NTOK * HDP * 2);
    bf16_t* vbt   = (bf16_t*)alloc((size_t)B_ * NHEAD * HDP * NTOK * 2);
    float*  sc    = (float*) alloc((size_t)B_ * NHEAD * NQP * NTOK * 4);
    bf16_t* pb    = (bf16_t*)alloc((size_t)B_ * NHEAD * NQP * NTOK * 2);
    float*  attn  = (float*) alloc((size_t)B_ * NQ * CDIM * 4);
    bf16_t* attnb = (bf16_t*)alloc((size_t)B_ * NQP * KPAD * 2);

    zero_kernel<<<1024, 256, 0, stream>>>((uint32_t*)wkv, (size_t)NKVP * KPAD * 2 / 4);
    zero_kernel<<<4096, 256, 0, stream>>>((uint32_t*)kb,
        ((size_t)B_ * NHEAD * NTOK * HDP * 2 * 2) / 4);            // kb + vbt contiguous
    zero_kernel<<<256, 256, 0, stream>>>((uint32_t*)attn, (size_t)B_ * NQ * CDIM);

    cvt_pad_kernel<<<4096, 256, 0, stream>>>(x,   xb,  B_ * NTOK, CDIM, KPAD);
    cvt_pad_kernel<<<512, 256, 0, stream>>>(kvw, wkv, NKV,       CDIM, KPAD);
    cvt_pad_kernel<<<512, 256, 0, stream>>>(pw,  wpj, CDIM,      CDIM, KPAD);
    qb_prep_kernel<<<256, 256, 0, stream>>>(qe, qb);

    kv_gemm_kernel<<<dim3(NKVP / 64, (B_ * NTOK) / 64), 256, 0, stream>>>(xb, wkv, kvb, kb, vbt);

    scores_kernel<<<dim3(NTOK / 1024, B_ * NHEAD), 256, 0, stream>>>(qb, kb, sc);

    softmax_kernel<<<B_ * NHEAD * NQP, 256, 0, stream>>>(sc, mask, pb);

    av_kernel<<<dim3(NTOK / 1024, B_ * NHEAD), 256, 0, stream>>>(pb, vbt, attn);

    attnb_prep_kernel<<<512, 256, 0, stream>>>(attn, attnb);
    int proj_waves = ((B_ * NQP) / 16) * (CDIM / 16);               // 462
    proj_gemm_kernel<<<(proj_waves + 7) / 8, 256, 0, stream>>>(attnb, wpj, pbias, out);
}
